// MambaBlock_28424093565333
// MI455X (gfx1250) — matrix operand order repaired and checked by reading, not compiled, not run on hardware
//
#include <hip/hip_runtime.h>
#include <math.h>

typedef __attribute__((ext_vector_type(16))) __bf16 v16bf;
typedef __attribute__((ext_vector_type(8)))  float  v8f;

#define L_SEQ     2048
#define DMODEL    1024
#define DINNER    2048
#define DSTATE    16
#define DTRANK    64
#define NPROJ     4096   // 2*DINNER
#define NXDBL     96     // DTRANK + 2*DSTATE

// ---------------------------------------------------------------------------
// f32 -> bf16 conversion
// ---------------------------------------------------------------------------
__global__ __launch_bounds__(256) void k_cvt_bf16(const float* __restrict__ src,
                                                  __bf16* __restrict__ dst, int n) {
  int i = blockIdx.x * 256 + threadIdx.x;
  if (i < n) dst[i] = (__bf16)src[i];
}

// ---------------------------------------------------------------------------
// WMMA GEMM:  C[M,N] (f32) = sum_k A[M,K] * B[N,K]   (A,B bf16 row-major)
// Requirements (hold at all call sites): M % 128 == 0, K % 32 == 0.
// N may be ragged (row-guarded).  Block tile 128x128, K-step 32.
// 256 threads = 8 waves (2 M x 4 N); each wave: 64x32 tile = 4x2 WMMA frags
// (8 wmma per wave per K-step, 12 ds_load_b128 -> 1.5 LDS loads per wmma).
// Staging is 16-byte vectorized and register-double-buffered.
// ---------------------------------------------------------------------------
#define BM 128
#define BN 128
#define BK 32
#define LDP (BK + 8)   // padded LDS row (bf16 elems); +16B keeps chunks aligned

__global__ __launch_bounds__(256) void k_gemm_bf16(
    const __bf16* __restrict__ A, const __bf16* __restrict__ B,
    float* __restrict__ C, int M, int N, int K, int lda, int ldb, int ldc) {
  __shared__ alignas(16) __bf16 sA[BM][LDP];   // 10 KB
  __shared__ alignas(16) __bf16 sB[BN][LDP];   // 10 KB

  const int tid  = threadIdx.x;
  const int lane = tid & 31;
  const int wave = tid >> 5;      // 0..7
  const int wm   = wave >> 2;     // 0..1  (M direction, 64 rows each)
  const int wn   = wave & 3;      // 0..3  (N direction, 32 cols each)
  const int half = lane >> 4;     // 0/1 lane half
  const int l16  = lane & 15;

  const int bm0 = blockIdx.y * BM;
  const int bn0 = blockIdx.x * BN;

  // A staging: 128x32 bf16 = 512 x 16B chunks, 2 per thread (4 chunks/row)
  const int aIdx0 = tid * 2, aIdx1 = tid * 2 + 1;
  const int ar0 = aIdx0 >> 2, ac0 = (aIdx0 & 3) * 8;
  const int ar1 = aIdx1 >> 2, ac1 = (aIdx1 & 3) * 8;
  // B staging: 128x32 bf16 = 512 x 16B chunks, 2 per thread
  const int br0 = ar0, bc0 = ac0;
  const int br1 = ar1, bc1 = ac1;
  const bool bIn0 = (bn0 + br0) < N;
  const bool bIn1 = (bn0 + br1) < N;

  const __bf16* Abase = A + (size_t)bm0 * lda;
  const __bf16* Bbase = B + (size_t)bn0 * ldb;

  uint4 aReg0, aReg1, bReg0, bReg1;
  auto loadTiles = [&](int k0) {
    aReg0 = *(const uint4*)(Abase + (size_t)ar0 * lda + k0 + ac0);
    aReg1 = *(const uint4*)(Abase + (size_t)ar1 * lda + k0 + ac1);
    bReg0 = bIn0 ? *(const uint4*)(Bbase + (size_t)br0 * ldb + k0 + bc0)
                 : uint4{0u, 0u, 0u, 0u};
    bReg1 = bIn1 ? *(const uint4*)(Bbase + (size_t)br1 * ldb + k0 + bc1)
                 : uint4{0u, 0u, 0u, 0u};
  };

  v8f acc[4][2] = {};
  loadTiles(0);

  for (int k0 = 0; k0 < K; k0 += BK) {
    __syncthreads();                 // previous tile's compute done with LDS
    *(uint4*)&sA[ar0][ac0] = aReg0;
    *(uint4*)&sA[ar1][ac1] = aReg1;
    *(uint4*)&sB[br0][bc0] = bReg0;
    *(uint4*)&sB[br1][bc1] = bReg1;
    __syncthreads();

    if (k0 + BK < K) {
      loadTiles(k0 + BK);            // overlap next-tile loads with WMMA
      if (k0 + 2 * BK < K) {         // warm L2 one tile further ahead
        __builtin_prefetch(Abase + (size_t)(tid & (BM - 1)) * lda + k0 + 2 * BK, 0, 3);
      }
    }

    // B fragments (shared across all 4 M tiles): lane -> N=l16, elem e -> K=e+16*half
    v16bf bfrag[2];
#pragma unroll
    for (int tn = 0; tn < 2; ++tn) {
      const __bf16* bp = &sB[wn * 32 + tn * 16 + l16][16 * half];
#pragma unroll
      for (int e = 0; e < 16; ++e) bfrag[tn][e] = bp[e];
    }

#pragma unroll
    for (int tm = 0; tm < 4; ++tm) {
      // A fragment: lane -> M=l16, elem e -> K=e+16*half
      v16bf afrag;
      const __bf16* ap = &sA[wm * 64 + tm * 16 + l16][0];
#pragma unroll
      for (int e = 0; e < 16; ++e) afrag[e] = ap[e + 16 * half];

#pragma unroll
      for (int tn = 0; tn < 2; ++tn) {
        acc[tm][tn] = __builtin_amdgcn_wmma_f32_16x16x32_bf16(
            false, afrag, false, bfrag[tn], (short)0, acc[tm][tn], false, false);
      }
    }
  }

  // store: C/D layout VGPR r -> M = r + 8*half, N = l16  (M always in range)
#pragma unroll
  for (int tm = 0; tm < 4; ++tm) {
#pragma unroll
    for (int tn = 0; tn < 2; ++tn) {
      int n = bn0 + wn * 32 + tn * 16 + l16;
      if (n < N) {
#pragma unroll
        for (int r = 0; r < 8; ++r) {
          int m = bm0 + wm * 64 + tm * 16 + r + 8 * half;
          C[(size_t)m * ldc + n] = acc[tm][tn][r];
        }
      }
    }
  }
}

// ---------------------------------------------------------------------------
// depthwise causal conv1d (taps=4, left pad 3) + bias + SiLU
// ---------------------------------------------------------------------------
__global__ __launch_bounds__(256) void k_conv_silu(
    const float* __restrict__ xandres, const float* __restrict__ conv_w,
    const float* __restrict__ conv_b, float* __restrict__ xi_f,
    __bf16* __restrict__ xi_bf) {
  int i = blockIdx.x * 256 + threadIdx.x;      // over L*DINNER
  int d = i & (DINNER - 1);
  int t = i >> 11;                             // /DINNER
  float acc = conv_b[d];
#pragma unroll
  for (int j = 0; j < 4; ++j) {
    int tt = t - 3 + j;
    if (tt >= 0) acc += conv_w[d * 4 + j] * xandres[(size_t)tt * NPROJ + d];
  }
  float s = acc / (1.f + __expf(-acc));        // SiLU
  xi_f[i]  = s;
  xi_bf[i] = (__bf16)s;
}

// extract dt-rank columns of x_dbl to a packed bf16 [L, 64] matrix
__global__ __launch_bounds__(256) void k_extract_dt(
    const float* __restrict__ xdbl, __bf16* __restrict__ dtbf) {
  int i = blockIdx.x * 256 + threadIdx.x;      // over L*DTRANK
  int t = i >> 6, c = i & 63;
  dtbf[i] = (__bf16)xdbl[t * NXDBL + c];
}

// in-place softplus(dt + b_dt)
__global__ __launch_bounds__(256) void k_softplus(
    float* __restrict__ dt, const float* __restrict__ b_dt) {
  int i = blockIdx.x * 256 + threadIdx.x;      // over L*DINNER
  float v = dt[i] + b_dt[i & (DINNER - 1)];
  dt[i] = (v > 20.f) ? v : log1pf(__expf(v));
}

// ---------------------------------------------------------------------------
// selective scan: one thread per channel d, h[16] state in VGPRs.
// B/C streams (shared by all d) staged through LDS in 128-timestep chunks.
// ---------------------------------------------------------------------------
#define TCH 128

__global__ __launch_bounds__(256) void k_scan(
    const float* __restrict__ delta, const float* __restrict__ xi_f,
    const float* __restrict__ xdbl,  const float* __restrict__ xandres,
    const float* __restrict__ A_log, const float* __restrict__ Dvec,
    __bf16* __restrict__ ybf) {
  __shared__ alignas(16) float sBC[TCH * 32];  // 16 KB: [t][B0..15,C0..15]
  const int tid = threadIdx.x;
  const int d   = blockIdx.x * 256 + tid;

  float Arow[DSTATE], h[DSTATE];
#pragma unroll
  for (int n = 0; n < DSTATE; ++n) {
    Arow[n] = -__expf(A_log[d * DSTATE + n]);
    h[n] = 0.f;
  }
  const float Dd = Dvec[d];

  for (int t0 = 0; t0 < L_SEQ; t0 += TCH) {
    // stage B/C: 32 f32 per timestep = 8 x float4 chunks per row (16B aligned)
    for (int i = tid; i < TCH * 8; i += 256) {
      int tt = i >> 3, c = (i & 7) * 4;
      *(float4*)&sBC[tt * 32 + c] =
          *(const float4*)&xdbl[(size_t)(t0 + tt) * NXDBL + DTRANK + c];
    }
    __syncthreads();

    for (int tt = 0; tt < TCH; ++tt) {
      const int t = t0 + tt;
      const float dlt = delta[(size_t)t * DINNER + d];
      const float xv  = xi_f[(size_t)t * DINNER + d];
      const float* bc = &sBC[tt * 32];
      float y = 0.f;
#pragma unroll
      for (int n = 0; n < DSTATE; ++n) {
        float dA = __expf(dlt * Arow[n]);
        h[n] = h[n] * dA + dlt * bc[n] * xv;
        y += h[n] * bc[16 + n];
      }
      y += xv * Dd;
      float r = xandres[(size_t)t * NPROJ + DINNER + d];
      y *= r / (1.f + __expf(-r));             // * silu(res)
      ybf[(size_t)t * DINNER + d] = (__bf16)y;
    }
    __syncthreads();
  }
}

// ---------------------------------------------------------------------------
extern "C" void kernel_launch(void* const* d_in, const int* in_sizes, int n_in,
                              void* d_out, int out_size, void* d_ws, size_t ws_size,
                              hipStream_t stream) {
  const float* x      = (const float*)d_in[0];
  const float* W_in   = (const float*)d_in[1];
  const float* conv_w = (const float*)d_in[2];
  const float* conv_b = (const float*)d_in[3];
  const float* W_x    = (const float*)d_in[4];
  const float* W_dt   = (const float*)d_in[5];
  const float* b_dt   = (const float*)d_in[6];
  const float* A_log  = (const float*)d_in[7];
  const float* Dvec   = (const float*)d_in[8];
  const float* W_out  = (const float*)d_in[9];
  float* out = (float*)d_out;

  char* w = (char*)d_ws;
  size_t off = 0;
  auto take = [&](size_t bytes) -> char* {
    char* p = w + off;
    off += (bytes + 255) & ~(size_t)255;
    return p;
  };
  __bf16* x_bf    = (__bf16*)take((size_t)L_SEQ * DMODEL * 2);
  __bf16* Win_bf  = (__bf16*)take((size_t)NPROJ * DMODEL * 2);
  __bf16* Wx_bf   = (__bf16*)take((size_t)NXDBL * DINNER * 2);
  __bf16* Wdt_bf  = (__bf16*)take((size_t)DINNER * DTRANK * 2);
  __bf16* Wout_bf = (__bf16*)take((size_t)DMODEL * DINNER * 2);
  float*  xandres = (float*) take((size_t)L_SEQ * NPROJ * 4);
  float*  xi_f    = (float*) take((size_t)L_SEQ * DINNER * 4);
  __bf16* xi_bf   = (__bf16*)take((size_t)L_SEQ * DINNER * 2);
  float*  xdbl    = (float*) take((size_t)L_SEQ * NXDBL * 4);
  __bf16* dt_bf   = (__bf16*)take((size_t)L_SEQ * DTRANK * 2);
  float*  dt_f    = (float*) take((size_t)L_SEQ * DINNER * 4);
  __bf16* y_bf    = (__bf16*)take((size_t)L_SEQ * DINNER * 2);

  auto cvt = [&](const float* s, __bf16* dptr, int n) {
    k_cvt_bf16<<<(n + 255) / 256, 256, 0, stream>>>(s, dptr, n);
  };
  cvt(x,     x_bf,    L_SEQ * DMODEL);
  cvt(W_in,  Win_bf,  NPROJ * DMODEL);
  cvt(W_x,   Wx_bf,   NXDBL * DINNER);
  cvt(W_dt,  Wdt_bf,  DINNER * DTRANK);
  cvt(W_out, Wout_bf, DMODEL * DINNER);

  auto gemm = [&](const __bf16* A, const __bf16* B, float* C,
                  int M, int N, int K, int lda, int ldb, int ldc) {
    dim3 g((N + BN - 1) / BN, (M + BM - 1) / BM);
    k_gemm_bf16<<<g, 256, 0, stream>>>(A, B, C, M, N, K, lda, ldb, ldc);
  };

  // 1) in_proj: [L,DM] x [NPROJ,DM]^T -> [L,NPROJ]
  gemm(x_bf, Win_bf, xandres, L_SEQ, NPROJ, DMODEL, DMODEL, DMODEL, NPROJ);
  // 2) depthwise conv + SiLU
  k_conv_silu<<<(L_SEQ * DINNER) / 256, 256, 0, stream>>>(
      xandres, conv_w, conv_b, xi_f, xi_bf);
  // 3) x_dbl: [L,DI] x [96,DI]^T -> [L,96]
  gemm(xi_bf, Wx_bf, xdbl, L_SEQ, NXDBL, DINNER, DINNER, DINNER, NXDBL);
  // 4) delta GEMM: [L,64] x [DI,64]^T -> [L,DI], then softplus(+b_dt)
  k_extract_dt<<<(L_SEQ * DTRANK) / 256, 256, 0, stream>>>(xdbl, dt_bf);
  gemm(dt_bf, Wdt_bf, dt_f, L_SEQ, DINNER, DTRANK, DTRANK, DTRANK, DINNER);
  k_softplus<<<(L_SEQ * DINNER) / 256, 256, 0, stream>>>(dt_f, b_dt);
  // 5) selective scan + skip + gate
  k_scan<<<DINNER / 256, 256, 0, stream>>>(dt_f, xi_f, xdbl, xandres,
                                           A_log, Dvec, y_bf);
  // 6) out_proj: [L,DI] x [DM,DI]^T -> [L,DM]  (writes d_out, f32)
  gemm(y_bf, Wout_bf, out, L_SEQ, DMODEL, DINNER, DINNER, DINNER, DMODEL);
}
